// clam_mil_11768210391321
// MI455X (gfx1250) — compile-verified, matrix operand order checked by reading, not run on hardware
//
#include <hip/hip_runtime.h>
#include <hip/hip_bf16.h>
#include <math.h>

typedef __attribute__((ext_vector_type(16))) __bf16 v16bf;
typedef __attribute__((ext_vector_type(8)))  __bf16 v8bf;
typedef __attribute__((ext_vector_type(8)))  float  v8f;
typedef __attribute__((__vector_size__(16))) int    i32x4;

#define BATCH 8
#define NINST 10000
#define NF    1024
#define ZD    512
#define NCLS  4

#if defined(__has_builtin)
#if __has_builtin(__builtin_amdgcn_global_store_async_from_lds_b128)
#define HAVE_ASYNC_LDS_STORE 1
#endif
#endif
#ifndef HAVE_ASYNC_LDS_STORE
#define HAVE_ASYNC_LDS_STORE 0
#endif

// ---------------------------------------------------------------------------
// Weight packing: f32 [K][Ncols] row-major  ->  bf16 WMMA B-fragment order:
// [K/32][Ncols/16][lane 0..31][16 bf16]  (32 contiguous bytes per lane).
// B-matrix 32x16 bf16 layout (wave32): lanes 0-15 col N=lane, K={0..7,16..23};
// lanes 16-31 col N=lane-16, K={8..15,24..31}; element e packs 2 bf16/VGPR.
// ---------------------------------------------------------------------------
__global__ void pack_weights_kernel(const float* __restrict__ src,
                                    __bf16* __restrict__ dst,
                                    int K, int Ncols) {
    int lane = threadIdx.x;                 // 0..31
    int nTilesN = Ncols >> 4;
    int kt = blockIdx.x / nTilesN;
    int ct = blockIdx.x % nTilesN;
    int k0 = kt * 32;
    int col = ct * 16 + (lane & 15);
    int hi  = lane >> 4;
    v16bf frag;
#pragma unroll
    for (int e = 0; e < 16; ++e) {
        int kk;
        if (e < 8) kk = k0 + (hi ? 8 : 0) + e;     // lanes<16: k0+e ; lanes>=16: k0+8+e
        else       kk = k0 + (hi ? 16 : 8) + e;    // lanes<16: 16..23 ; lanes>=16: 24..31
        frag[e] = (__bf16)src[(size_t)kk * Ncols + col];
    }
    *((v16bf*)dst + (size_t)blockIdx.x * 32 + lane) = frag;
}

// ---------------------------------------------------------------------------
// A-fragment (16x32 bf16) loader from row-major bf16 LDS tile with XOR swizzle
// on 16-byte chunks (chunk' = chunk ^ (row & 7)).
// Lane l<16: row=l, K={k0..k0+7, k0+16..k0+23}; lane>=16: K shifted by 8.
// ---------------------------------------------------------------------------
__device__ __forceinline__ v16bf load_afrag_lds(const char* base, int row,
                                                int rowStride, int k0, int lane) {
    int hi = lane >> 4;
    int sw = row & 7;
    int c1 = (k0 >> 3) + hi;
    int c2 = c1 + 2;
    const v8bf* p1 = (const v8bf*)(base + row * rowStride + ((c1 ^ sw) << 4));
    const v8bf* p2 = (const v8bf*)(base + row * rowStride + ((c2 ^ sw) << 4));
    v8bf lo = *p1;
    v8bf hh = *p2;
    v16bf r;
#pragma unroll
    for (int i = 0; i < 8; ++i) { r[i] = lo[i]; r[i + 8] = hh[i]; }
    return r;
}

// ---------------------------------------------------------------------------
// Pass 1 (fused): per 16-instance tile:
//   emb = relu(bags @ Wenc + b_enc)          (WMMA, K=1024)
//   u = sigmoid(emb @ Wu + b_u), v = tanh(emb @ Wv + b_v)   (WMMA, K=512)
//   logits[m][c] = sum_z (u*v)[m][z] * W_attn[c][z] + b_attn[c]
// emb written to HBM as bf16 for pass 3 (async LDS->global when available,
// overlapping the store with GEMM2; ASYNCcnt drains at S_ENDPGM).
// ---------------------------------------------------------------------------
__global__ void __launch_bounds__(256, 2)
mil_pass1_kernel(const float* __restrict__ bags,
                 const v16bf* __restrict__ pWenc, const float* __restrict__ b_enc,
                 const v16bf* __restrict__ pWu,   const float* __restrict__ b_u,
                 const v16bf* __restrict__ pWv,   const float* __restrict__ b_v,
                 const float* __restrict__ W_attn, const float* __restrict__ b_attn,
                 __bf16* __restrict__ embg, float* __restrict__ logitsg) {
    __shared__ __align__(16) char smem[49152];
    char*  bagsL = smem;               // [16][1024] bf16 swizzled = 32 KB
    char*  embL  = smem + 32768;       // [16][512]  bf16 swizzled = 16 KB
    float* partL = (float*)smem;       // [256 lanes][32] f32 = 32 KB (aliases bagsL)

    const int tid  = threadIdx.x;
    const int wave = tid >> 5;
    const int lane = tid & 31;
    const int hi   = lane >> 4;
    const int lcol = lane & 15;

    const int b    = blockIdx.x / (NINST / 16);
    const int tile = blockIdx.x % (NINST / 16);
    const int n0   = tile * 16;

    // ---- stage bags tile [16 x 1024] f32 -> bf16 LDS (swizzled 16B chunks) ----
    {
        const float4* src = (const float4*)(bags + ((size_t)b * NINST + n0) * NF);
        int row = tid >> 4;
        int sub = tid & 15;
#pragma unroll
        for (int i = 0; i < 8; ++i) {
            int cb = sub + (i << 4);                       // chunk of 8 elems
            float4 x = src[row * (NF / 4) + cb * 2];
            float4 y = src[row * (NF / 4) + cb * 2 + 1];
            v8bf v;
            v[0] = (__bf16)x.x; v[1] = (__bf16)x.y; v[2] = (__bf16)x.z; v[3] = (__bf16)x.w;
            v[4] = (__bf16)y.x; v[5] = (__bf16)y.y; v[6] = (__bf16)y.z; v[7] = (__bf16)y.w;
            *(v8bf*)(bagsL + row * 2048 + ((cb ^ (row & 7)) << 4)) = v;
        }
    }
    __syncthreads();

    // ---- GEMM1: encoder, wave owns Z columns [wave*64, wave*64+64) ----
    v8f acc[4];
#pragma unroll
    for (int ct = 0; ct < 4; ++ct)
#pragma unroll
        for (int i = 0; i < 8; ++i) acc[ct][i] = 0.0f;

    for (int kt = 0; kt < NF / 32; ++kt) {
        v16bf a = load_afrag_lds(bagsL, lcol, 2048, kt * 32, lane);
#pragma unroll
        for (int ct = 0; ct < 4; ++ct) {
            int gct = wave * 4 + ct;
            v16bf bf = pWenc[((size_t)kt * (ZD / 16) + gct) * 32 + lane];
            acc[ct] = __builtin_amdgcn_wmma_f32_16x16x32_bf16(
                false, a, false, bf, (short)0, acc[ct], false, false);
        }
    }

    // bias + relu -> emb LDS (bf16, swizzled).  C layout: row m = r + hi*8,
    // col = tile_col_base + (lane&15).
#pragma unroll
    for (int ct = 0; ct < 4; ++ct) {
        int col  = wave * 64 + ct * 16 + lcol;
        float bias = b_enc[col];
        int cc   = col >> 3;
        int cofs = (col & 7) * 2;
#pragma unroll
        for (int r = 0; r < 8; ++r) {
            int m = r + hi * 8;
            float x = fmaxf(acc[ct][r] + bias, 0.0f);
            *(__bf16*)(embL + m * 1024 + ((cc ^ (m & 7)) << 4) + cofs) = (__bf16)x;
        }
    }
    __syncthreads();

    // ---- spill emb tile LDS -> HBM bf16 [b][n][z] ----
    // Async LDS->global (ASYNCcnt) overlaps the spill with GEMM2 below; embL is
    // read-only for the remainder of the block, and S_ENDPGM drains ASYNCcnt.
#pragma unroll
    for (int i = 0; i < 4; ++i) {
        int ci  = tid + i * 256;                            // 0..1023 chunks
        int row = ci >> 6;
        int cb  = ci & 63;
        char* lsrc = embL + row * 1024 + ((cb ^ (row & 7)) << 4);
        char* gdst = (char*)embg + (((size_t)b * NINST + n0 + row) * ZD + cb * 8) * 2;
#if HAVE_ASYNC_LDS_STORE
        __builtin_amdgcn_global_store_async_from_lds_b128(
            (__attribute__((address_space(1))) i32x4*)gdst,
            (__attribute__((address_space(3))) i32x4*)lsrc, 0, 0);
#else
        *(uint4*)gdst = *(const uint4*)lsrc;
#endif
    }

    // ---- GEMM2: U and V (A = emb tile from LDS, K=512) ----
    v8f accU[4], accV[4];
#pragma unroll
    for (int ct = 0; ct < 4; ++ct)
#pragma unroll
        for (int i = 0; i < 8; ++i) { accU[ct][i] = 0.0f; accV[ct][i] = 0.0f; }

    for (int kt = 0; kt < ZD / 32; ++kt) {
        v16bf a = load_afrag_lds(embL, lcol, 1024, kt * 32, lane);
#pragma unroll
        for (int ct = 0; ct < 4; ++ct) {
            int gct = wave * 4 + ct;
            v16bf bu = pWu[((size_t)kt * (ZD / 16) + gct) * 32 + lane];
            v16bf bv = pWv[((size_t)kt * (ZD / 16) + gct) * 32 + lane];
            accU[ct] = __builtin_amdgcn_wmma_f32_16x16x32_bf16(
                false, a, false, bu, (short)0, accU[ct], false, false);
            accV[ct] = __builtin_amdgcn_wmma_f32_16x16x32_bf16(
                false, a, false, bv, (short)0, accV[ct], false, false);
        }
    }

    // ---- gating + per-lane attention-logit partials ----
    float part[NCLS][8];
#pragma unroll
    for (int c = 0; c < NCLS; ++c)
#pragma unroll
        for (int r = 0; r < 8; ++r) part[c][r] = 0.0f;

#pragma unroll
    for (int ct = 0; ct < 4; ++ct) {
        int col = wave * 64 + ct * 16 + lcol;
        float bu = b_u[col], bv = b_v[col];
        float wa0 = W_attn[0 * ZD + col];
        float wa1 = W_attn[1 * ZD + col];
        float wa2 = W_attn[2 * ZD + col];
        float wa3 = W_attn[3 * ZD + col];
#pragma unroll
        for (int r = 0; r < 8; ++r) {
            float uu = 1.0f / (1.0f + __expf(-(accU[ct][r] + bu)));
            float vv = tanhf(accV[ct][r] + bv);
            float g  = uu * vv;
            part[0][r] += g * wa0;
            part[1][r] += g * wa1;
            part[2][r] += g * wa2;
            part[3][r] += g * wa3;
        }
    }

    // deterministic cross-wave reduction (partL aliases bagsL; GEMM1 is done)
    float* myPart = partL + (size_t)(wave * 32 + lane) * 32;
#pragma unroll
    for (int c = 0; c < NCLS; ++c)
#pragma unroll
        for (int r = 0; r < 8; ++r) myPart[c * 8 + r] = part[c][r];
    __syncthreads();

    if (tid < 64) {
        int m = tid >> 2, c = tid & 3;
        int r = m & 7, hh = m >> 3;
        float s = 0.0f;
        for (int w = 0; w < 8; ++w)
            for (int l = 0; l < 16; ++l)
                s += partL[(size_t)((w * 32 + hh * 16 + l)) * 32 + c * 8 + r];
        logitsg[((size_t)b * NINST + n0 + m) * NCLS + c] = s + b_attn[c];
    }
}

// ---------------------------------------------------------------------------
// Pass 2: per (b,c) softmax max & sum-exp over N instances.
// ---------------------------------------------------------------------------
__global__ void mil_stats_kernel(const float* __restrict__ logitsg,
                                 float* __restrict__ statsM,
                                 float* __restrict__ statsS) {
    __shared__ float red[256];
    int b = blockIdx.x >> 2, c = blockIdx.x & 3;
    int t = threadIdx.x;
    float m = -1e30f;
    for (int n = t; n < NINST; n += 256)
        m = fmaxf(m, logitsg[((size_t)b * NINST + n) * NCLS + c]);
    red[t] = m; __syncthreads();
    for (int s = 128; s > 0; s >>= 1) {
        if (t < s) red[t] = fmaxf(red[t], red[t + s]);
        __syncthreads();
    }
    float mx = red[0]; __syncthreads();
    float su = 0.0f;
    for (int n = t; n < NINST; n += 256)
        su += __expf(logitsg[((size_t)b * NINST + n) * NCLS + c] - mx);
    red[t] = su; __syncthreads();
    for (int s = 128; s > 0; s >>= 1) {
        if (t < s) red[t] += red[t + s];
        __syncthreads();
    }
    if (t == 0) { statsM[blockIdx.x] = mx; statsS[blockIdx.x] = red[0]; }
}

// ---------------------------------------------------------------------------
// Pass 3: reps[b][c][z] = sum_n softmax_w[b][n][c] * emb[b][n][z]
// One block per (b, z-half); softmax weights staged per 128-instance chunk.
// ---------------------------------------------------------------------------
__global__ void __launch_bounds__(256)
mil_reps_kernel(const float* __restrict__ logitsg, const __bf16* __restrict__ embg,
                const float* __restrict__ statsM, const float* __restrict__ statsS,
                float* __restrict__ reps) {
    __shared__ float wts[128 * NCLS];
    int b = blockIdx.x >> 1;
    int z = (blockIdx.x & 1) * 256 + threadIdx.x;
    float mx[NCLS], inv[NCLS];
#pragma unroll
    for (int c = 0; c < NCLS; ++c) {
        mx[c]  = statsM[b * NCLS + c];
        inv[c] = 1.0f / statsS[b * NCLS + c];
    }
    float acc[NCLS] = {0.0f, 0.0f, 0.0f, 0.0f};
    for (int nb = 0; nb < NINST; nb += 128) {
        int cnt = min(128, NINST - nb);
        __syncthreads();
        for (int v = threadIdx.x; v < cnt * NCLS; v += 256) {
            int nn = v >> 2, c = v & 3;
            wts[v] = __expf(logitsg[((size_t)b * NINST + nb + nn) * NCLS + c] - mx[c]) * inv[c];
        }
        __syncthreads();
        const __bf16* ep = embg + ((size_t)b * NINST + nb) * ZD + z;
        for (int nn = 0; nn < cnt; ++nn) {
            float e = (float)ep[(size_t)nn * ZD];
#pragma unroll
            for (int c = 0; c < NCLS; ++c) acc[c] += wts[nn * NCLS + c] * e;
        }
    }
#pragma unroll
    for (int c = 0; c < NCLS; ++c)
        reps[((size_t)b * NCLS + c) * ZD + z] = acc[c];
}

// ---------------------------------------------------------------------------
// Pass 4: scores[b][c] = dot(reps[b][c], W_cls[c]) + b_cls[c]
// ---------------------------------------------------------------------------
__global__ void mil_scores_kernel(const float* __restrict__ reps,
                                  const float* __restrict__ W_cls,
                                  const float* __restrict__ b_cls,
                                  float* __restrict__ out) {
    __shared__ float red[64];
    int b = blockIdx.x >> 2, c = blockIdx.x & 3;
    int t = threadIdx.x;
    float s = 0.0f;
    for (int i = t; i < ZD; i += 64)
        s += reps[((size_t)b * NCLS + c) * ZD + i] * W_cls[c * ZD + i];
    red[t] = s; __syncthreads();
    for (int k = 32; k > 0; k >>= 1) {
        if (t < k) red[t] += red[t + k];
        __syncthreads();
    }
    if (t == 0) out[b * NCLS + c] = red[0] + b_cls[c];
}

// ---------------------------------------------------------------------------
extern "C" void kernel_launch(void* const* d_in, const int* in_sizes, int n_in,
                              void* d_out, int out_size, void* d_ws, size_t ws_size,
                              hipStream_t stream) {
    const float* bags   = (const float*)d_in[0];
    const float* W_enc  = (const float*)d_in[1];
    const float* b_enc  = (const float*)d_in[2];
    const float* W_u    = (const float*)d_in[3];
    const float* b_u    = (const float*)d_in[4];
    const float* W_v    = (const float*)d_in[5];
    const float* b_v    = (const float*)d_in[6];
    const float* W_attn = (const float*)d_in[7];
    const float* b_attn = (const float*)d_in[8];
    const float* W_cls  = (const float*)d_in[9];
    const float* b_cls  = (const float*)d_in[10];

    char* ws = (char*)d_ws;
    size_t off = 0;
    auto carve = [&](size_t bytes) -> char* {
        char* p = ws + off;
        off += (bytes + 255) & ~(size_t)255;
        return p;
    };
    __bf16* pWenc  = (__bf16*)carve((size_t)NF * ZD * 2);
    __bf16* pWu    = (__bf16*)carve((size_t)ZD * ZD * 2);
    __bf16* pWv    = (__bf16*)carve((size_t)ZD * ZD * 2);
    __bf16* embg   = (__bf16*)carve((size_t)BATCH * NINST * ZD * 2);
    float*  logitsg = (float*)carve((size_t)BATCH * NINST * NCLS * 4);
    float*  statsM  = (float*)carve((size_t)BATCH * NCLS * 4);
    float*  statsS  = (float*)carve((size_t)BATCH * NCLS * 4);
    float*  repsg   = (float*)carve((size_t)BATCH * NCLS * ZD * 4);

    pack_weights_kernel<<<(NF / 32) * (ZD / 16), 32, 0, stream>>>(W_enc, pWenc, NF, ZD);
    pack_weights_kernel<<<(ZD / 32) * (ZD / 16), 32, 0, stream>>>(W_u, pWu, ZD, ZD);
    pack_weights_kernel<<<(ZD / 32) * (ZD / 16), 32, 0, stream>>>(W_v, pWv, ZD, ZD);

    mil_pass1_kernel<<<BATCH * (NINST / 16), 256, 0, stream>>>(
        bags, (const v16bf*)pWenc, b_enc, (const v16bf*)pWu, b_u,
        (const v16bf*)pWv, b_v, W_attn, b_attn, embg, logitsg);

    mil_stats_kernel<<<BATCH * NCLS, 256, 0, stream>>>(logitsg, statsM, statsS);
    mil_reps_kernel<<<BATCH * 2, 256, 0, stream>>>(logitsg, embg, statsM, statsS, repsg);
    mil_scores_kernel<<<BATCH * NCLS, 64, 0, stream>>>(repsg, W_cls, b_cls, (float*)d_out);
}
